// EncCore_88613765251690
// MI455X (gfx1250) — compile-verified, matrix operand order checked by reading, not run on hardware
//
#include <hip/hip_runtime.h>

// ---------------------------------------------------------------------------
// EncNet head on MI455X (gfx1250): all GEMMs via v_wmma_f32_16x16x32_bf16,
// with double-buffered async Global->LDS staging (ASYNCcnt) in the two heavy
// convolutions. Compute-bound (~618 GFLOP in the 3x3 conv) -> bf16 WMMA.
// ---------------------------------------------------------------------------

typedef __attribute__((ext_vector_type(16))) __bf16 v16bf;
typedef __attribute__((ext_vector_type(8)))  float  v8f;

#define INV_BN 0.9999950000374997f   // 1/sqrt(1 + 1e-5)

constexpr int K_   = 32;
constexpr int OUT_ = 150;
constexpr int NPAD = 160;            // tail N padded to 10 wmma tiles

// ---- workspace layout (bytes; total ~255 MB) ----
constexpr size_t OFF_XB    = 0;                        // bf16 x, NHWC   [8][64][64][2048]
constexpr size_t OFF_WTB   = OFF_XB    + 134217728;    // bf16 w_top    [9][64][512][32]
constexpr size_t OFF_WEB   = OFF_WTB   + 18874368;     // bf16 w_enc    [16][512][32]
constexpr size_t OFF_TAILB = OFF_WEB   + 524288;       // bf16 tail_w   [16][160][32]
constexpr size_t OFF_HB    = OFF_TAILB + 163840;       // bf16 h, chan-last [32768][512]
constexpr size_t OFF_Z     = OFF_HB    + 33554432;     // f32  Xf       [32768][512]
constexpr size_t OFF_E     = OFF_Z     + 67108864;     // f32  E        [8][32][512]
constexpr size_t OFF_ASUM  = OFF_E     + 524288;       // f32  Asum     [8][32]
constexpr size_t OFF_ZBUF  = OFF_ASUM  + 1024;         // f32  zeros    [16] (pad source)
constexpr size_t OFF_CODE  = OFF_ZBUF  + 64;           // f32  code     [8][512]
constexpr size_t OFF_GATE  = OFF_CODE  + 16384;        // f32  gate     [8][512]

union FragU { v16bf v; float4 f4[2]; };
union H8    { float4 f4; __bf16 h[8]; };

// CDNA5 async tensor path: 16B Global -> LDS, tracked by ASYNCcnt (§15.18.3).
// LDS address = low 32 bits of the flat shared pointer (LDS aperture rule).
__device__ __forceinline__ void async_ld16(const __bf16* lds_dst, const void* gsrc) {
  unsigned lds_addr = (unsigned)(size_t)lds_dst;
  asm volatile("global_load_async_to_lds_b128 %0, %1, off"
               :: "v"(lds_addr), "v"(gsrc) : "memory");
}
__device__ __forceinline__ void wait_async0() {
  asm volatile("s_wait_asynccnt 0" ::: "memory");
}

// Load a 16-bit A/B WMMA fragment from an LDS row (pitch-40 bf16, 16B aligned).
// Per ISA 7.12.2: lane<16 holds K 0..7 & 16..23, lane>=16 holds K 8..15 & 24..31.
__device__ __forceinline__ v16bf frag_ld(const __bf16* rowBase, int hk) {
  FragU u;
  u.f4[0] = *(const float4*)(rowBase + 8 * hk);
  u.f4[1] = *(const float4*)(rowBase + 16 + 8 * hk);
  return u.v;
}

// ---------------------------------------------------------------------------
// Prep kernels: f32 -> bf16, layout transforms so GEMM staging is contiguous.
// ---------------------------------------------------------------------------
__global__ void prep_x(const float* __restrict__ x, __bf16* __restrict__ xb) {
  size_t i = (size_t)blockIdx.x * 256 + threadIdx.x;   // output idx [b][y][x][c]
  const int c  = (int)(i & 2047);
  const size_t s = i >> 11;                            // b*4096 + hw
  const int b  = (int)(s >> 12);
  const int hw = (int)(s & 4095);
  xb[i] = (__bf16)x[((size_t)b * 2048 + c) * 4096 + hw];
}

__global__ void prep_wtop(const float* __restrict__ w, __bf16* __restrict__ wtb) {
  size_t i = (size_t)blockIdx.x * 256 + threadIdx.x;   // [t][cb][n][kk]
  const int kk = (int)(i & 31);
  const int n  = (int)((i >> 5) & 511);
  const int cb = (int)((i >> 14) & 63);
  const int t  = (int)(i >> 20);
  wtb[i] = (__bf16)w[((size_t)n * 2048 + cb * 32 + kk) * 9 + t];
}

__global__ void prep_wenc(const float* __restrict__ w, __bf16* __restrict__ web) {
  int i = blockIdx.x * 256 + threadIdx.x;              // [cb][n][kk]
  const int kk = i & 31;
  const int n  = (i >> 5) & 511;
  const int cb = i >> 14;
  web[i] = (__bf16)w[(size_t)n * 512 + cb * 32 + kk];
}

__global__ void prep_wtail(const float* __restrict__ w, __bf16* __restrict__ tb) {
  int i = blockIdx.x * 256 + threadIdx.x;              // [cb][n(160)][kk]
  const int kk = i & 31;
  const int n  = (i >> 5) % NPAD;
  const int cb = (i >> 5) / NPAD;
  tb[i] = (n < OUT_) ? (__bf16)w[(size_t)n * 512 + cb * 32 + kk] : (__bf16)0.0f;
}

__global__ void zero_f32(float* p, int n) {
  int i = blockIdx.x * 256 + threadIdx.x;
  if (i < n) p[i] = 0.0f;
}

// ---------------------------------------------------------------------------
// Conv 3x3 (implicit GEMM, M=32768 N=512 K=18432) + BN + ReLU -> hb (bf16 NHWC)
// Double-buffered async Global->LDS pipeline, one barrier per K-step.
// ---------------------------------------------------------------------------
__global__ __launch_bounds__(256) void conv3x3_wmma(
    const __bf16* __restrict__ xb, const __bf16* __restrict__ wtb,
    const __bf16* __restrict__ zpad,
    const float* __restrict__ g1, const float* __restrict__ b1,
    __bf16* __restrict__ hb) {
  __shared__ __bf16 Asm[2][128 * 40];
  __shared__ __bf16 Bsm[2][64 * 40];
  const int tid = threadIdx.x;
  const int lane = tid & 31, wave = tid >> 5;
  const int wm = wave & 3, wn = wave >> 2;
  const int m0 = blockIdx.x * 128;
  const int n0 = blockIdx.y * 64;
  const int hk = lane >> 4, l15 = lane & 15;
  v8f acc[2][2] = {};
  constexpr int NT = 9 * 64;

  auto issue_tile = [&](int it, int p) {
    const int t = it >> 6, cb = it & 63;
    const int dy = t / 3 - 1, dx = t % 3 - 1;
    const int cin0 = cb * 32;
    // A: 128 rows x 32 bf16 = 512 x 16B chunks, 2 per thread (OOB -> zero page)
#pragma unroll
    for (int cc = 0; cc < 2; ++cc) {
      const int c = tid * 2 + cc;
      const int r = c >> 2, q = c & 3;
      const int m = m0 + r;
      const int b = m >> 12, hw = m & 4095;
      const int sy = (hw >> 6) + dy, sx = (hw & 63) + dx;
      const bool ok = ((unsigned)sy < 64u) && ((unsigned)sx < 64u);
      const __bf16* g = ok
          ? (xb + ((((size_t)b * 64 + sy) * 64 + sx) * 2048 + cin0 + q * 8))
          : zpad;
      async_ld16(&Asm[p][r * 40 + q * 8], g);
    }
    // B: 64 rows x 32 bf16 = 256 x 16B chunks, 1 per thread
    {
      const int n = tid >> 2, q = tid & 3;
      async_ld16(&Bsm[p][n * 40 + q * 8],
                 wtb + ((size_t)(t * 64 + cb) * 512 + n0 + n) * 32 + q * 8);
    }
  };

  issue_tile(0, 0);
  for (int it = 0; it < NT; ++it) {
    const int p = it & 1;
    wait_async0();          // this wave's tile-it data resident in LDS
    __syncthreads();        // all waves have tile it; all done reading buf[1-p]
    if (it + 1 < NT) issue_tile(it + 1, p ^ 1);
    v16bf af[2], bfm[2];
#pragma unroll
    for (int i = 0; i < 2; ++i)
      af[i] = frag_ld(&Asm[p][(wm * 32 + i * 16 + l15) * 40], hk);
#pragma unroll
    for (int j = 0; j < 2; ++j)
      bfm[j] = frag_ld(&Bsm[p][(wn * 32 + j * 16 + l15) * 40], hk);
#pragma unroll
    for (int i = 0; i < 2; ++i)
#pragma unroll
      for (int j = 0; j < 2; ++j)
        acc[i][j] = __builtin_amdgcn_wmma_f32_16x16x32_bf16(
            false, af[i], false, bfm[j], (short)0, acc[i][j], false, false);
  }
  // fused BN + ReLU epilogue, channel-last bf16 store
#pragma unroll
  for (int i = 0; i < 2; ++i) {
#pragma unroll
    for (int j = 0; j < 2; ++j) {
      const int n = n0 + wn * 32 + j * 16 + l15;
      const float g = g1[n] * INV_BN, bb = b1[n];
#pragma unroll
      for (int v = 0; v < 8; ++v) {
        const int m = m0 + wm * 32 + i * 16 + v + 8 * hk;
        float hv = g * acc[i][j][v] + bb;
        hv = hv > 0.f ? hv : 0.f;
        hb[(size_t)m * 512 + n] = (__bf16)hv;
      }
    }
  }
}

// ---------------------------------------------------------------------------
// Conv 1x1 enc (GEMM M=32768 N=512 K=512) + BN + ReLU -> z = Xf (f32 chan-last)
// Same async double-buffered pipeline.
// ---------------------------------------------------------------------------
__global__ __launch_bounds__(256) void conv1x1_wmma(
    const __bf16* __restrict__ hb, const __bf16* __restrict__ web,
    const float* __restrict__ g2, const float* __restrict__ b2,
    float* __restrict__ z) {
  __shared__ __bf16 Asm[2][128 * 40];
  __shared__ __bf16 Bsm[2][64 * 40];
  const int tid = threadIdx.x;
  const int lane = tid & 31, wave = tid >> 5;
  const int wm = wave & 3, wn = wave >> 2;
  const int m0 = blockIdx.x * 128;
  const int n0 = blockIdx.y * 64;
  const int hk = lane >> 4, l15 = lane & 15;
  v8f acc[2][2] = {};
  constexpr int NT = 16;

  auto issue_tile = [&](int cb, int p) {
#pragma unroll
    for (int cc = 0; cc < 2; ++cc) {
      const int c = tid * 2 + cc;
      const int r = c >> 2, q = c & 3;
      async_ld16(&Asm[p][r * 40 + q * 8],
                 hb + (size_t)(m0 + r) * 512 + cb * 32 + q * 8);
    }
    {
      const int n = tid >> 2, q = tid & 3;
      async_ld16(&Bsm[p][n * 40 + q * 8],
                 web + ((size_t)cb * 512 + n0 + n) * 32 + q * 8);
    }
  };

  issue_tile(0, 0);
  for (int it = 0; it < NT; ++it) {
    const int p = it & 1;
    wait_async0();
    __syncthreads();
    if (it + 1 < NT) issue_tile(it + 1, p ^ 1);
    v16bf af[2], bfm[2];
#pragma unroll
    for (int i = 0; i < 2; ++i)
      af[i] = frag_ld(&Asm[p][(wm * 32 + i * 16 + l15) * 40], hk);
#pragma unroll
    for (int j = 0; j < 2; ++j)
      bfm[j] = frag_ld(&Bsm[p][(wn * 32 + j * 16 + l15) * 40], hk);
#pragma unroll
    for (int i = 0; i < 2; ++i)
#pragma unroll
      for (int j = 0; j < 2; ++j)
        acc[i][j] = __builtin_amdgcn_wmma_f32_16x16x32_bf16(
            false, af[i], false, bfm[j], (short)0, acc[i][j], false, false);
  }
#pragma unroll
  for (int i = 0; i < 2; ++i) {
#pragma unroll
    for (int j = 0; j < 2; ++j) {
      const int n = n0 + wn * 32 + j * 16 + l15;
      const float g = g2[n] * INV_BN, bb = b2[n];
#pragma unroll
      for (int v = 0; v < 8; ++v) {
        const int m = m0 + wm * 32 + i * 16 + v + 8 * hk;
        float zv = g * acc[i][j][v] + bb;
        z[(size_t)m * 512 + n] = zv > 0.f ? zv : 0.f;
      }
    }
  }
}

// ---------------------------------------------------------------------------
// Encoding: per pixel-row softmax over K=32 codewords (lane=k, wave32 shuffle
// reductions), residual aggregation into bank-swizzled LDS, atomic flush to E.
// ---------------------------------------------------------------------------
__global__ __launch_bounds__(256) void enc_aggregate(
    const float* __restrict__ z, const float* __restrict__ cw,
    const float* __restrict__ scale, float* __restrict__ E,
    float* __restrict__ Asum) {
  __shared__ float Elds[K_ * 512];   // swizzled: [d][k] -> 64KB, conflict-free
  const int tid = threadIdx.x;
  const int k = tid & 31, wave = tid >> 5;
  const int b = blockIdx.y;
  for (int e = tid; e < K_ * 512; e += 256) Elds[e] = 0.f;
  __syncthreads();

  float c2k = 0.f;
  for (int d = 0; d < 512; ++d) {
    float cv = cw[(size_t)k * 512 + d];
    c2k += cv * cv;
  }
  const float sk = scale[k];
  float asum_l = 0.f;

  for (int rr = 0; rr < 16; ++rr) {
    const int n = blockIdx.x * 128 + wave * 16 + rr;
    const float* Xrow = z + ((size_t)b * 4096 + n) * 512;
    float dot = 0.f, x2 = 0.f;
    for (int d = 0; d < 512; ++d) {
      const float xv = Xrow[d];
      dot += xv * cw[(size_t)k * 512 + d];
      x2  += xv * xv;
    }
    float sl = sk * (x2 + c2k - 2.f * dot);
    float mx = sl;
#pragma unroll
    for (int off = 16; off > 0; off >>= 1) mx = fmaxf(mx, __shfl_xor(mx, off, 32));
    float a = __expf(sl - mx);
    float s = a;
#pragma unroll
    for (int off = 16; off > 0; off >>= 1) s += __shfl_xor(s, off, 32);
    a /= s;
    asum_l += a;
    for (int d = 0; d < 512; ++d)
      atomicAdd(&Elds[d * 32 + k], a * Xrow[d]);
  }
  __syncthreads();
  atomicAdd(&Asum[b * 32 + k], asum_l);
  for (int e = tid; e < K_ * 512; e += 256)
    atomicAdd(&E[((size_t)b * 32 + (e & 31)) * 512 + (e >> 5)], Elds[e]);
}

// E -> residual -> BN(K) -> ReLU -> mean over K  => code[b][d]
__global__ void enc_finalize(const float* __restrict__ E, const float* __restrict__ Asum,
                             const float* __restrict__ cw, const float* __restrict__ g3,
                             const float* __restrict__ b3, float* __restrict__ code) {
  const int idx = blockIdx.x * 256 + threadIdx.x;   // 8*512
  const int b = idx >> 9, d = idx & 511;
  float s = 0.f;
  for (int k = 0; k < K_; ++k) {
    float e = E[((size_t)b * 32 + k) * 512 + d] - Asum[b * 32 + k] * cw[(size_t)k * 512 + d];
    float v = g3[k] * INV_BN * e + b3[k];
    s += v > 0.f ? v : 0.f;
  }
  code[idx] = s * (1.0f / 32.0f);
}

// gate = sigmoid(code @ fc_w^T + fc_b)
__global__ void gate_fc(const float* __restrict__ code, const float* __restrict__ fw,
                        const float* __restrict__ fb, float* __restrict__ gate) {
  const int idx = blockIdx.x * 256 + threadIdx.x;   // 8*512
  const int b = idx >> 9, o = idx & 511;
  float s = fb[o];
  for (int d = 0; d < 512; ++d)
    s += code[b * 512 + d] * fw[(size_t)o * 512 + d];
  gate[idx] = 1.0f / (1.0f + __expf(-s));
}

// ---------------------------------------------------------------------------
// Tail: A = relu(h*(1+gate)) generated on the fly in bf16; GEMM M=32768 K=512
// N=160 (o>=150 masked). seg[b][o][hw] = acc + tail_b[o].
// ---------------------------------------------------------------------------
__global__ __launch_bounds__(256) void tail_wmma(
    const __bf16* __restrict__ hb, const __bf16* __restrict__ tb,
    const float* __restrict__ gate, const float* __restrict__ tbias,
    float* __restrict__ seg) {
  __shared__ __bf16 Alds[128 * 40];
  __shared__ __bf16 Blds[NPAD * 40];
  __shared__ float  glds[512];
  const int tid = threadIdx.x;
  const int lane = tid & 31, wave = tid >> 5;
  const int wm = wave & 3, wn = wave >> 2;     // 4 x 2 waves -> 128M x 160N tile
  const int m0 = blockIdx.x * 128;
  const int b  = m0 >> 12;                     // 128 | 4096 -> single batch per block
  const int hk = lane >> 4, l15 = lane & 15;
  glds[tid]       = 1.0f + gate[b * 512 + tid];
  glds[tid + 256] = 1.0f + gate[b * 512 + tid + 256];
  __syncthreads();
  v8f acc[2][5] = {};

  for (int cb = 0; cb < 16; ++cb) {
    const int cin0 = cb * 32;
#pragma unroll
    for (int cc = 0; cc < 2; ++cc) {
      const int c = tid * 2 + cc;
      const int r = c >> 2, q = c & 3;
      H8 in, outv;
      in.f4 = *(const float4*)(hb + (size_t)(m0 + r) * 512 + cin0 + q * 8);
#pragma unroll
      for (int e = 0; e < 8; ++e) {
        float vv = (float)in.h[e] * glds[cin0 + q * 8 + e];
        outv.h[e] = (__bf16)(vv > 0.f ? vv : 0.f);
      }
      *(float4*)(Alds + r * 40 + q * 8) = outv.f4;
    }
    for (int c = tid; c < NPAD * 4; c += 256) {
      const int n = c >> 2, q = c & 3;
      *(float4*)(Blds + n * 40 + q * 8) = *(const float4*)(
          tb + ((size_t)cb * NPAD + n) * 32 + q * 8);
    }
    __syncthreads();
    v16bf af[2], bfm[5];
#pragma unroll
    for (int i = 0; i < 2; ++i)
      af[i] = frag_ld(Alds + (wm * 32 + i * 16 + l15) * 40, hk);
#pragma unroll
    for (int j = 0; j < 5; ++j)
      bfm[j] = frag_ld(Blds + (wn * 80 + j * 16 + l15) * 40, hk);
#pragma unroll
    for (int i = 0; i < 2; ++i)
#pragma unroll
      for (int j = 0; j < 5; ++j)
        acc[i][j] = __builtin_amdgcn_wmma_f32_16x16x32_bf16(
            false, af[i], false, bfm[j], (short)0, acc[i][j], false, false);
    __syncthreads();
  }
#pragma unroll
  for (int i = 0; i < 2; ++i) {
#pragma unroll
    for (int j = 0; j < 5; ++j) {
      const int o = wn * 80 + j * 16 + l15;
      if (o < OUT_) {
        const float bias = tbias[o];
#pragma unroll
        for (int v = 0; v < 8; ++v) {
          const int m = m0 + wm * 32 + i * 16 + v + 8 * hk;
          const int hw = m & 4095;
          seg[((size_t)b * OUT_ + o) * 4096 + hw] = acc[i][j][v] + bias;
        }
      }
    }
  }
}

// ---------------------------------------------------------------------------
extern "C" void kernel_launch(void* const* d_in, const int* in_sizes, int n_in,
                              void* d_out, int out_size, void* d_ws, size_t ws_size,
                              hipStream_t stream) {
  (void)in_sizes; (void)n_in; (void)out_size; (void)ws_size;
  const float* x      = (const float*)d_in[0];
  const float* w_top  = (const float*)d_in[1];
  const float* g1     = (const float*)d_in[2];
  const float* b1     = (const float*)d_in[3];
  const float* w_enc  = (const float*)d_in[4];
  const float* g2     = (const float*)d_in[5];
  const float* b2     = (const float*)d_in[6];
  const float* cw     = (const float*)d_in[7];
  const float* scale  = (const float*)d_in[8];
  const float* g3     = (const float*)d_in[9];
  const float* b3     = (const float*)d_in[10];
  const float* fc_w   = (const float*)d_in[11];
  const float* fc_b   = (const float*)d_in[12];
  const float* tail_w = (const float*)d_in[13];
  const float* tail_b = (const float*)d_in[14];
  float* seg = (float*)d_out;

  char* ws = (char*)d_ws;
  __bf16* xb    = (__bf16*)(ws + OFF_XB);
  __bf16* wtb   = (__bf16*)(ws + OFF_WTB);
  __bf16* web   = (__bf16*)(ws + OFF_WEB);
  __bf16* tailb = (__bf16*)(ws + OFF_TAILB);
  __bf16* hb    = (__bf16*)(ws + OFF_HB);
  float*  z     = (float*)(ws + OFF_Z);
  float*  E     = (float*)(ws + OFF_E);
  float*  Asum  = (float*)(ws + OFF_ASUM);
  __bf16* zpad  = (__bf16*)(ws + OFF_ZBUF);
  float*  code  = (float*)(ws + OFF_CODE);
  float*  gate  = (float*)(ws + OFF_GATE);

  // prep; zero E + Asum + zero-pad page in one pass (contiguous region)
  prep_x    <<<262144, 256, 0, stream>>>(x, xb);
  prep_wtop <<<36864,  256, 0, stream>>>(w_top, wtb);
  prep_wenc <<<1024,   256, 0, stream>>>(w_enc, web);
  prep_wtail<<<320,    256, 0, stream>>>(tail_w, tailb);
  zero_f32  <<<(131072 + 256 + 16 + 255) / 256, 256, 0, stream>>>(E, 131072 + 256 + 16);

  // heavy GEMMs (async double-buffered WMMA)
  conv3x3_wmma<<<dim3(256, 8), 256, 0, stream>>>(xb, wtb, zpad, g1, b1, hb);
  conv1x1_wmma<<<dim3(256, 8), 256, 0, stream>>>(hb, web, g2, b2, z);

  // encoding + gate
  enc_aggregate<<<dim3(32, 8), 256, 0, stream>>>(z, cw, scale, E, Asum);
  enc_finalize <<<16, 256, 0, stream>>>(E, Asum, cw, g3, b3, code);
  gate_fc      <<<16, 256, 0, stream>>>(code, fc_w, fc_b, gate);

  // gated tail GEMM -> seg
  tail_wmma<<<256, 256, 0, stream>>>(hb, tailb, gate, tail_b, seg);
}
